// TransformerWithConditionalAttention_12103217840441
// MI455X (gfx1250) — matrix operand order repaired and checked by reading, not compiled, not run on hardware
//
#include <hip/hip_runtime.h>
#include <math.h>

// ---------------- problem constants ----------------
#define B_   4
#define N_   4096
#define D_   512
#define H_   8
#define DH_  64
#define NQ_  1024
#define NKV_ 1024
#define L_   2
#define E_   4
#define EPS_f 0.1f
#define NITER_ 20

typedef __bf16 bf16;
typedef __attribute__((ext_vector_type(16))) __bf16 v16bf;
typedef __attribute__((ext_vector_type(8)))  __bf16 v8bf;
typedef __attribute__((ext_vector_type(8)))  float  v8f;

__device__ __forceinline__ v16bf cat8(v8bf lo, v8bf hi) {
  return __builtin_shufflevector(lo, hi, 0, 1, 2, 3, 4, 5, 6, 7, 8, 9, 10, 11, 12, 13, 14, 15);
}

// =====================================================================
// Router gate logits: s[b,n] = dot(x[b,n,:], w)   (one wave per token)
// =====================================================================
__global__ __launch_bounds__(256) void k_router(const float* __restrict__ x,
                                                const float* __restrict__ wqv,
                                                const float* __restrict__ wkv,
                                                float* __restrict__ sq,
                                                float* __restrict__ skv) {
  int wave = (blockIdx.x * blockDim.x + threadIdx.x) >> 5;
  int lane = threadIdx.x & 31;
  if (wave >= B_ * N_) return;
  const float* xr = x + (size_t)wave * D_;
  float aq = 0.f, ak = 0.f;
  for (int j = lane; j < D_; j += 32) {
    float xv = xr[j];
    aq += xv * wqv[j];
    ak += xv * wkv[j];
  }
  for (int m = 16; m >= 1; m >>= 1) {
    aq += __shfl_xor(aq, m, 32);
    ak += __shfl_xor(ak, m, 32);
  }
  if (lane == 0) { sq[wave] = aq; skv[wave] = ak; }
}

// =====================================================================
// Coordinate-descent soft top-k scores (one block per batch row)
// =====================================================================
__global__ __launch_bounds__(256) void k_coor(const float* __restrict__ s_in,
                                              float* __restrict__ scores,
                                              int kval) {
  __shared__ float red[256];
  int b = blockIdx.x, tid = threadIdx.x;
  const float* s = s_in + (size_t)b * N_;
  float sv[16], bv[16];
  for (int i = 0; i < 16; i++) { sv[i] = s[tid + i * 256]; bv[i] = -sv[i]; }
  float logk = logf((float)kval);
  float a = 0.f;
  for (int it = 0; it < NITER_; it++) {
    float lm = -INFINITY;
    for (int i = 0; i < 16; i++) lm = fmaxf(lm, sv[i] + bv[i]);
    red[tid] = lm; __syncthreads();
    for (int off = 128; off > 0; off >>= 1) { if (tid < off) red[tid] = fmaxf(red[tid], red[tid + off]); __syncthreads(); }
    float M = red[0] / EPS_f; __syncthreads();
    float ls = 0.f;
    for (int i = 0; i < 16; i++) ls += __expf((sv[i] + bv[i]) / EPS_f - M);
    red[tid] = ls; __syncthreads();
    for (int off = 128; off > 0; off >>= 1) { if (tid < off) red[tid] += red[tid + off]; __syncthreads(); }
    float lse = M + logf(red[0]); __syncthreads();
    a = EPS_f * (logk - lse);
    for (int i = 0; i < 16; i++) bv[i] = -fmaxf(sv[i] + a, 0.f);
  }
  for (int i = 0; i < 16; i++)
    scores[(size_t)b * N_ + tid + i * 256] = __expf((sv[i] + a + bv[i]) / EPS_f);
}

// =====================================================================
// Top-k (k = 1024 of 4096) via in-LDS bitonic sort, descending
// =====================================================================
__global__ __launch_bounds__(512) void k_topk(const float* __restrict__ scores,
                                              int* __restrict__ idx_out,
                                              float* __restrict__ sc_out) {
  __shared__ float key[N_];
  __shared__ int   idx[N_];
  int b = blockIdx.x, tid = threadIdx.x, nt = blockDim.x;
  for (int i = tid; i < N_; i += nt) { key[i] = scores[(size_t)b * N_ + i]; idx[i] = i; }
  __syncthreads();
  for (int k = 2; k <= N_; k <<= 1) {
    for (int j = k >> 1; j > 0; j >>= 1) {
      for (int t = tid; t < N_; t += nt) {
        int ixj = t ^ j;
        if (ixj > t) {
          bool up = ((t & k) == 0);
          float kt = key[t], kx = key[ixj];
          if ((kt < kx) == up) {
            key[t] = kx; key[ixj] = kt;
            int it = idx[t]; idx[t] = idx[ixj]; idx[ixj] = it;
          }
        }
      }
      __syncthreads();
    }
  }
  for (int i = tid; i < NQ_; i += nt) {
    idx_out[b * NQ_ + i] = idx[i];
    sc_out[b * NQ_ + i]  = key[i];
  }
}

// =====================================================================
// Gather routed tokens + LayerNorm -> bf16 rows (GEMM A operand)
// =====================================================================
__global__ __launch_bounds__(256) void k_gather_ln_bf(const float* __restrict__ x,
                                                      const int* __restrict__ idx,
                                                      const float* __restrict__ gam,
                                                      const float* __restrict__ bet,
                                                      bf16* __restrict__ out) {
  __shared__ float red[256];
  int row = blockIdx.x;            // b*NQ + i
  int b = row / NQ_;
  int tok = idx[row];
  const float* xr = x + ((size_t)b * N_ + tok) * D_;
  int tid = threadIdx.x;
  float v0 = xr[tid], v1 = xr[tid + 256];
  red[tid] = v0 + v1; __syncthreads();
  for (int off = 128; off > 0; off >>= 1) { if (tid < off) red[tid] += red[tid + off]; __syncthreads(); }
  float mu = red[0] / (float)D_; __syncthreads();
  float d0 = v0 - mu, d1 = v1 - mu;
  red[tid] = d0 * d0 + d1 * d1; __syncthreads();
  for (int off = 128; off > 0; off >>= 1) { if (tid < off) red[tid] += red[tid + off]; __syncthreads(); }
  float rs = rsqrtf(red[0] / (float)D_ + 1e-5f);
  bf16* orow = out + (size_t)row * D_;
  orow[tid]       = (bf16)(d0 * rs * gam[tid] + bet[tid]);
  orow[tid + 256] = (bf16)(d1 * rs * gam[tid + 256] + bet[tid + 256]);
}

// Row LayerNorm -> bf16 (FFN input)
__global__ __launch_bounds__(256) void k_ln_bf(const float* __restrict__ in,
                                               const float* __restrict__ gam,
                                               const float* __restrict__ bet,
                                               bf16* __restrict__ out) {
  __shared__ float red[256];
  int row = blockIdx.x;
  const float* xr = in + (size_t)row * D_;
  int tid = threadIdx.x;
  float v0 = xr[tid], v1 = xr[tid + 256];
  red[tid] = v0 + v1; __syncthreads();
  for (int off = 128; off > 0; off >>= 1) { if (tid < off) red[tid] += red[tid + off]; __syncthreads(); }
  float mu = red[0] / (float)D_; __syncthreads();
  float d0 = v0 - mu, d1 = v1 - mu;
  red[tid] = d0 * d0 + d1 * d1; __syncthreads();
  for (int off = 128; off > 0; off >>= 1) { if (tid < off) red[tid] += red[tid + off]; __syncthreads(); }
  float rs = rsqrtf(red[0] / (float)D_ + 1e-5f);
  bf16* orow = out + (size_t)row * D_;
  orow[tid]       = (bf16)(d0 * rs * gam[tid] + bet[tid]);
  orow[tid + 256] = (bf16)(d1 * rs * gam[tid + 256] + bet[tid + 256]);
}

// Row LayerNorm -> f32 (final output)
__global__ __launch_bounds__(256) void k_ln_f32(const float* __restrict__ in,
                                                const float* __restrict__ gam,
                                                const float* __restrict__ bet,
                                                float* __restrict__ out) {
  __shared__ float red[256];
  int row = blockIdx.x;
  const float* xr = in + (size_t)row * D_;
  int tid = threadIdx.x;
  float v0 = xr[tid], v1 = xr[tid + 256];
  red[tid] = v0 + v1; __syncthreads();
  for (int off = 128; off > 0; off >>= 1) { if (tid < off) red[tid] += red[tid + off]; __syncthreads(); }
  float mu = red[0] / (float)D_; __syncthreads();
  float d0 = v0 - mu, d1 = v1 - mu;
  red[tid] = d0 * d0 + d1 * d1; __syncthreads();
  for (int off = 128; off > 0; off >>= 1) { if (tid < off) red[tid] += red[tid + off]; __syncthreads(); }
  float rs = rsqrtf(red[0] / (float)D_ + 1e-5f);
  float* orow = out + (size_t)row * D_;
  orow[tid]       = d0 * rs * gam[tid] + bet[tid];
  orow[tid + 256] = d1 * rs * gam[tid + 256] + bet[tid + 256];
}

// =====================================================================
// Weight prep: f32 B[K][N] -> bf16 BT[N][K]  (once per launch)
// =====================================================================
__global__ __launch_bounds__(256) void k_transpose_bf(const float* __restrict__ Bsrc,
                                                      bf16* __restrict__ BT,
                                                      int K, int N) {
  int t = blockIdx.x * blockDim.x + threadIdx.x;
  if (t >= K * N) return;
  int k = t / N, n = t - k * N;
  BT[(size_t)n * K + k] = (bf16)Bsrc[t];
}

// =====================================================================
// GEMM: C(MxN,f32) = A(MxK,bf16 row-major) @ BT(NxK,bf16 row-major)^T
// one wave -> 16x32 C strip (2 tiles sharing the A fragment)
// =====================================================================
__global__ __launch_bounds__(256) void k_gemm_bf16(const bf16* __restrict__ A,
                                                   const bf16* __restrict__ BT,
                                                   float* __restrict__ C,
                                                   int M, int K, int Nc) {
  int wid  = (int)((blockIdx.x * blockDim.x + threadIdx.x) >> 5);
  int lane = threadIdx.x & 31;
  int tilesN = Nc >> 5;                 // 32 columns per wave
  int tot = (M >> 4) * tilesN;
  if (wid >= tot) return;
  int tm = wid / tilesN, tn = wid % tilesN;
  int m = lane & 15, g = lane >> 4;
  const bf16* arow  = A  + (size_t)(tm * 16 + m) * K;
  const bf16* brow0 = BT + (size_t)(tn * 32 + m) * K;
  const bf16* brow1 = brow0 + (size_t)16 * K;
  v8f acc0, acc1;
  for (int r = 0; r < 8; r++) { acc0[r] = 0.f; acc1[r] = 0.f; }
  for (int k0 = 0; k0 < K; k0 += 32) {
    __builtin_prefetch(brow0 + k0 + 64, 0, 1);   // global_prefetch_b8
    v16bf afr = cat8(*(const v8bf*)(arow + k0 + g * 16),
                     *(const v8bf*)(arow + k0 + g * 16 + 8));
    v16bf bf0 = cat8(*(const v8bf*)(brow0 + k0 + g * 16),
                     *(const v8bf*)(brow0 + k0 + g * 16 + 8));
    v16bf bf1 = cat8(*(const v8bf*)(brow1 + k0 + g * 16),
                     *(const v8bf*)(brow1 + k0 + g * 16 + 8));
    acc0 = __builtin_amdgcn_wmma_f32_16x16x32_bf16(false, afr, false, bf0, (short)0, acc0, false, false);
    acc1 = __builtin_amdgcn_wmma_f32_16x16x32_bf16(false, afr, false, bf1, (short)0, acc1, false, false);
  }
  for (int r = 0; r < 8; r++) {
    size_t orow = (size_t)(tm * 16 + r + 8 * g) * Nc + tn * 32;
    C[orow + m]      = acc0[r];
    C[orow + 16 + m] = acc1[r];
  }
}

// =====================================================================
// Q post: scale 1/sqrt(DH), rotary(rotary_emb[qi]) -> bf16 [b][h][i][d]
// =====================================================================
__global__ __launch_bounds__(256) void k_qpost(const float* __restrict__ qlin,
                                               const int* __restrict__ qi,
                                               const float* __restrict__ rot,
                                               bf16* __restrict__ qbf) {
  int t = blockIdx.x * blockDim.x + threadIdx.x;
  if (t >= B_ * NQ_ * D_) return;
  int d512 = t & 511; int row = t >> 9;       // row = b*NQ + i
  int h = d512 >> 6, d = d512 & 63;
  int b = row >> 10, i = row & 1023;
  const float sc = 0.125f;                     // DH^-0.5
  float v = qlin[t] * sc;
  int partner = (d < 32) ? (d512 + 32) : (d512 - 32);
  float vp = qlin[(size_t)row * 512 + partner] * sc;
  float rh = (d < 32) ? -vp : vp;
  float f = rot[(size_t)qi[row] * DH_ + d];
  float o = v * __cosf(f) + rh * __sinf(f);
  qbf[(((size_t)(b * H_ + h) * NQ_) + i) * DH_ + d] = (bf16)o;
}

// =====================================================================
// KV post: k -> rotary -> bf16 [b][h][j][d] ; v -> *kv_sc -> bf16 [b][h][d][j]
// =====================================================================
__global__ __launch_bounds__(256) void k_kvpost(const float* __restrict__ kv,
                                                const int* __restrict__ ki,
                                                const float* __restrict__ kvsc,
                                                const float* __restrict__ rot,
                                                bf16* __restrict__ kb,
                                                bf16* __restrict__ vT) {
  int t = blockIdx.x * blockDim.x + threadIdx.x;
  if (t >= B_ * NKV_ * D_) return;
  int d512 = t & 511; int row = t >> 9;
  int h = d512 >> 6, d = d512 & 63;
  int b = row >> 10, i = row & 1023;
  const float* kvr = kv + (size_t)row * 1024;
  float kd = kvr[d512];
  int partner = (d < 32) ? (d512 + 32) : (d512 - 32);
  float kp = kvr[partner];
  float rh = (d < 32) ? -kp : kp;
  float f = rot[(size_t)ki[row] * DH_ + d];
  float kval = kd * __cosf(f) + rh * __sinf(f);
  kb[(((size_t)(b * H_ + h) * NKV_) + i) * DH_ + d] = (bf16)kval;
  float vval = kvr[512 + d512] * kvsc[row];
  vT[(((size_t)(b * H_ + h) * DH_) + d) * NKV_ + i] = (bf16)vval;
}

// =====================================================================
// Flash attention: wave owns 16 query rows of one (b,h); online softmax;
// all operands bf16 with fragment-order contiguous 16B loads
// =====================================================================
__global__ __launch_bounds__(256) void k_attn(const bf16* __restrict__ qbf,
                                              const bf16* __restrict__ kb,
                                              const bf16* __restrict__ vT,
                                              bf16* __restrict__ olin) {
  __shared__ __bf16 Pt[8][16 * 32];
  int wavein = threadIdx.x >> 5, lane = threadIdx.x & 31;
  int wid = blockIdx.x * 8 + wavein;     // 0..2047 = B*H*(NQ/16)
  int itile = wid & 63;
  int bh = wid >> 6;
  int b = bh >> 3, h = bh & 7;
  int m = lane & 15, g = lane >> 4;

  const bf16* qrow = qbf + (((size_t)bh * NQ_) + itile * 16 + m) * DH_;
  v16bf aq0 = cat8(*(const v8bf*)(qrow + g * 16),      *(const v8bf*)(qrow + g * 16 + 8));
  v16bf aq1 = cat8(*(const v8bf*)(qrow + 32 + g * 16), *(const v8bf*)(qrow + 32 + g * 16 + 8));

  float mr[8], lr[8];
  v8f o0, o1, o2, o3;
  for (int r = 0; r < 8; r++) {
    mr[r] = -INFINITY; lr[r] = 0.f;
    o0[r] = 0.f; o1[r] = 0.f; o2[r] = 0.f; o3[r] = 0.f;
  }
  __bf16* P = &Pt[wavein][0];
  const bf16* kbh = kb + (size_t)bh * NKV_ * DH_;
  const bf16* vbh = vT + (size_t)bh * DH_ * NKV_;

  for (int jt = 0; jt < NKV_; jt += 32) {
    v8f s0, s1;
    for (int r = 0; r < 8; r++) { s0[r] = 0.f; s1[r] = 0.f; }
    const bf16* kr0 = kbh + (size_t)(jt + m) * DH_;        // key row j = jt+m
    const bf16* kr1 = kbh + (size_t)(jt + 16 + m) * DH_;   // key row j = jt+16+m
    v16bf bk;
    bk = cat8(*(const v8bf*)(kr0 + g * 16), *(const v8bf*)(kr0 + g * 16 + 8));
    s0 = __builtin_amdgcn_wmma_f32_16x16x32_bf16(false, aq0, false, bk, (short)0, s0, false, false);
    bk = cat8(*(const v8bf*)(kr0 + 32 + g * 16), *(const v8bf*)(kr0 + 32 + g * 16 + 8));
    s0 = __builtin_amdgcn_wmma_f32_16x16x32_bf16(false, aq1, false, bk, (short)0, s0, false, false);
    bk = cat8(*(const v8bf*)(kr1 + g * 16), *(const v8bf*)(kr1 + g * 16 + 8));
    s1 = __builtin_amdgcn_wmma_f32_16x16x32_bf16(false, aq0, false, bk, (short)0, s1, false, false);
    bk = cat8(*(const v8bf*)(kr1 + 32 + g * 16), *(const v8bf*)(kr1 + 32 + g * 16 + 8));
    s1 = __builtin_amdgcn_wmma_f32_16x16x32_bf16(false, aq1, false, bk, (short)0, s1, false, false);

    // online softmax per row (row = r + 8g); reductions within 16-lane group
    for (int r = 0; r < 8; r++) {
      float tmax = fmaxf(s0[r], s1[r]);
      for (int msk = 1; msk < 16; msk <<= 1) tmax = fmaxf(tmax, __shfl_xor(tmax, msk, 32));
      float mn = fmaxf(mr[r], tmax);
      float alpha = __expf(mr[r] - mn);
      mr[r] = mn;
      float p0 = __expf(s0[r] - mn), p1 = __expf(s1[r] - mn);
      float rs = p0 + p1;
      for (int msk = 1; msk < 16; msk <<= 1) rs += __shfl_xor(rs, msk, 32);
      lr[r] = lr[r] * alpha + rs;
      o0[r] *= alpha; o1[r] *= alpha; o2[r] *= alpha; o3[r] *= alpha;
      P[(r + 8 * g) * 32 + m]      = (__bf16)p0;
      P[(r + 8 * g) * 32 + 16 + m] = (__bf16)p1;
    }
    // P: C layout -> A-fragment layout via LDS
    v16bf ap = cat8(*(const v8bf*)(P + m * 32 + g * 16),
                    *(const v8bf*)(P + m * 32 + g * 16 + 8));
    // O += P(16x32) x V(32x16) per 16-wide d block; vT rows contiguous in j
    v16bf bvv;
    const bf16* vr0 = vbh + (size_t)m * NKV_ + jt;
    bvv = cat8(*(const v8bf*)(vr0 + g * 16), *(const v8bf*)(vr0 + g * 16 + 8));
    o0 = __builtin_amdgcn_wmma_f32_16x16x32_bf16(false, ap, false, bvv, (short)0, o0, false, false);
    const bf16* vr1 = vbh + (size_t)(16 + m) * NKV_ + jt;
    bvv = cat8(*(const v8bf*)(vr1 + g * 16), *(const v8bf*)(vr1 + g * 16 + 8));
    o1 = __builtin_amdgcn_wmma_f32_16x16x32_bf16(false, ap, false, bvv, (short)0, o1, false, false);
    const bf16* vr2 = vbh + (size_t)(32 + m) * NKV_ + jt;
    bvv = cat8(*(const v8bf*)(vr2 + g * 16), *(const v8bf*)(vr2 + g * 16 + 8));
    o2 = __builtin_amdgcn_wmma_f32_16x16x32_bf16(false, ap, false, bvv, (short)0, o2, false, false);
    const bf16* vr3 = vbh + (size_t)(48 + m) * NKV_ + jt;
    bvv = cat8(*(const v8bf*)(vr3 + g * 16), *(const v8bf*)(vr3 + g * 16 + 8));
    o3 = __builtin_amdgcn_wmma_f32_16x16x32_bf16(false, ap, false, bvv, (short)0, o3, false, false);
  }
  // epilogue: divide by l, write bf16 o_lin[b*NQ+i][h*64+d]
  for (int r = 0; r < 8; r++) {
    float inv = 1.f / lr[r];
    size_t orow = ((size_t)b * NQ_ + itile * 16 + r + 8 * g) * 512 + (size_t)h * 64;
    olin[orow + m]      = (bf16)(o0[r] * inv);
    olin[orow + 16 + m] = (bf16)(o1[r] * inv);
    olin[orow + 32 + m] = (bf16)(o2[r] * inv);
    olin[orow + 48 + m] = (bf16)(o3[r] * inv);
  }
}

// =====================================================================
// Scatter gated outputs into the expert-sum accumulator
// =====================================================================
__global__ __launch_bounds__(256) void k_scatter(const float* __restrict__ oproj,
                                                 const int* __restrict__ qi,
                                                 const float* __restrict__ qsc,
                                                 const float* __restrict__ nullt,
                                                 float* __restrict__ acc) {
  int t = blockIdx.x * blockDim.x + threadIdx.x;
  if (t >= B_ * NQ_ * D_) return;
  int d = t & 511; int row = t >> 9; int b = row >> 10;
  int tok = qi[row];
  float v = oproj[t] * qsc[row] - nullt[d];
  size_t a = ((size_t)b * N_ + tok) * D_ + d;
  acc[a] += v;
}

__global__ __launch_bounds__(256) void k_acc_init(const float* __restrict__ nulltok,
                                                  float* __restrict__ acc) {
  size_t t = (size_t)blockIdx.x * blockDim.x + threadIdx.x;
  if (t >= (size_t)B_ * N_ * D_) return;
  int d = (int)(t & 511);
  acc[t] = nulltok[d] + nulltok[512 + d] + nulltok[1024 + d] + nulltok[1536 + d];
}

__global__ __launch_bounds__(256) void k_combine(float* __restrict__ xcur,
                                                 const float* __restrict__ acc) {
  size_t t = (size_t)blockIdx.x * blockDim.x + threadIdx.x;
  if (t >= (size_t)B_ * N_ * D_) return;
  xcur[t] += 0.25f * acc[t];
}

// bias + tanh-GELU, emits bf16 operand for the second FF GEMM
__global__ __launch_bounds__(256) void k_bias_gelu_bf(const float* __restrict__ h,
                                                      const float* __restrict__ bias,
                                                      bf16* __restrict__ out) {
  size_t t = (size_t)blockIdx.x * blockDim.x + threadIdx.x;
  if (t >= (size_t)B_ * N_ * D_) return;
  int d = (int)(t & 511);
  float x = h[t] + bias[d];
  float u = 0.7978845608028654f * (x + 0.044715f * x * x * x);
  out[t] = (bf16)(0.5f * x * (1.f + tanhf(u)));
}

__global__ __launch_bounds__(256) void k_bias_add_res(float* __restrict__ xcur,
                                                      const float* __restrict__ h2,
                                                      const float* __restrict__ bias) {
  size_t t = (size_t)blockIdx.x * blockDim.x + threadIdx.x;
  if (t >= (size_t)B_ * N_ * D_) return;
  int d = (int)(t & 511);
  xcur[t] += h2[t] + bias[d];
}

// =====================================================================
// Host orchestration
// =====================================================================
extern "C" void kernel_launch(void* const* d_in, const int* in_sizes, int n_in,
                              void* d_out, int out_size, void* d_ws, size_t ws_size,
                              hipStream_t stream) {
  (void)in_sizes; (void)n_in; (void)out_size; (void)ws_size;
  const float* x     = (const float*)d_in[0];
  const float* rot   = (const float*)d_in[1];
  const float* rqw   = (const float*)d_in[2];
  const float* rkvw  = (const float*)d_in[3];
  const float* lnqg  = (const float*)d_in[4];
  const float* lnqb  = (const float*)d_in[5];
  const float* lncg  = (const float*)d_in[6];
  const float* lncb  = (const float*)d_in[7];
  const float* wq    = (const float*)d_in[8];
  const float* wkv   = (const float*)d_in[9];
  const float* wo    = (const float*)d_in[10];
  const float* nullt = (const float*)d_in[11];
  const float* ffg   = (const float*)d_in[12];
  const float* ffb   = (const float*)d_in[13];
  const float* ffw1  = (const float*)d_in[14];
  const float* ffb1  = (const float*)d_in[15];
  const float* ffw2  = (const float*)d_in[16];
  const float* ffb2  = (const float*)d_in[17];
  const float* outg  = (const float*)d_in[18];
  const float* outb  = (const float*)d_in[19];

  const size_t MB1 = 1024ull * 1024ull;
  const size_t BND = (size_t)B_ * N_ * D_;      // 8388608
  size_t off = 0;
  char* wsb = (char*)d_ws;
  auto ALLOC = [&](size_t n) { char* p = wsb + off; off += (n + 255) & ~(size_t)255; return p; };

  float* xcur = (float*)ALLOC(BND * 4);
  float* acc  = (float*)ALLOC(BND * 4);
  float* sq   = (float*)ALLOC((size_t)B_ * N_ * 4);
  float* skv  = (float*)ALLOC((size_t)B_ * N_ * 4);
  float* scq  = (float*)ALLOC((size_t)B_ * N_ * 4);
  float* sckv = (float*)ALLOC((size_t)B_ * N_ * 4);
  int*   qi   = (int*)ALLOC((size_t)B_ * NQ_ * 4);
  int*   ki   = (int*)ALLOC((size_t)B_ * NQ_ * 4);
  float* qsc  = (float*)ALLOC((size_t)B_ * NQ_ * 4);
  float* kvsc = (float*)ALLOC((size_t)B_ * NQ_ * 4);
  // pre-transposed bf16 weights (persistent for the launch)
  bf16* wqT   = (bf16*)ALLOC((size_t)L_ * E_ * D_ * D_ * 2);
  bf16* wkvT  = (bf16*)ALLOC((size_t)L_ * E_ * D_ * 2 * D_ * 2);
  bf16* woT   = (bf16*)ALLOC((size_t)L_ * E_ * D_ * D_ * 2);
  bf16* ffw1T = (bf16*)ALLOC((size_t)L_ * D_ * D_ * 2);
  bf16* ffw2T = (bf16*)ALLOC((size_t)L_ * D_ * D_ * 2);
  char* pool  = ALLOC(64 * MB1);   // union region, reused across phases
  // expert phase
  bf16*  qnb   = (bf16*)(pool);
  bf16*  ctxb  = (bf16*)(pool + 4 * MB1);
  float* qlin  = (float*)(pool + 8 * MB1);
  float* kvb   = (float*)(pool + 16 * MB1);
  bf16*  qbf   = (bf16*)(pool + 32 * MB1);
  bf16*  kbbuf = (bf16*)(pool + 36 * MB1);
  bf16*  vTbuf = (bf16*)(pool + 40 * MB1);
  bf16*  olinb = (bf16*)(pool + 44 * MB1);
  float* oproj = (float*)(pool + 48 * MB1);
  // FF phase reuse
  bf16*  hlnb  = (bf16*)(pool);
  float* h1    = (float*)(pool + 16 * MB1);
  bf16*  h1b   = (bf16*)(pool + 48 * MB1);

  hipMemcpyAsync(xcur, x, BND * 4, hipMemcpyDeviceToDevice, stream);

  // ---- weight prep (bf16, transposed to [N][K]) ----
  for (int le = 0; le < L_ * E_; le++) {
    k_transpose_bf<<<1024, 256, 0, stream>>>(wq  + (size_t)le * D_ * D_,     wqT  + (size_t)le * D_ * D_,     D_, D_);
    k_transpose_bf<<<2048, 256, 0, stream>>>(wkv + (size_t)le * D_ * 2 * D_, wkvT + (size_t)le * D_ * 2 * D_, D_, 2 * D_);
    k_transpose_bf<<<1024, 256, 0, stream>>>(wo  + (size_t)le * D_ * D_,     woT  + (size_t)le * D_ * D_,     D_, D_);
  }
  for (int l = 0; l < L_; l++) {
    k_transpose_bf<<<1024, 256, 0, stream>>>(ffw1 + (size_t)l * D_ * D_, ffw1T + (size_t)l * D_ * D_, D_, D_);
    k_transpose_bf<<<1024, 256, 0, stream>>>(ffw2 + (size_t)l * D_ * D_, ffw2T + (size_t)l * D_ * D_, D_, D_);
  }

  const int EW_BLK = 256;
  const int EW_GRID_BND  = (int)(BND / EW_BLK);                   // 32768
  const int EW_GRID_ROUT = (int)((size_t)B_ * NQ_ * D_ / EW_BLK); // 8192

  for (int l = 0; l < L_; l++) {
    k_acc_init<<<EW_GRID_BND, EW_BLK, 0, stream>>>(nullt + (size_t)l * E_ * D_, acc);

    for (int e = 0; e < E_; e++) {
      size_t le = (size_t)l * E_ + e;
      const float* rq_w  = rqw  + le * D_;
      const float* rk_w  = rkvw + le * D_;
      const float* qg    = lnqg + le * D_;
      const float* qb    = lnqb + le * D_;
      const float* cg    = lncg + le * D_;
      const float* cb    = lncb + le * D_;
      const bf16*  wqTe  = wqT  + le * (size_t)D_ * D_;
      const bf16*  wkvTe = wkvT + le * (size_t)D_ * 2 * D_;
      const bf16*  woTe  = woT  + le * (size_t)D_ * D_;
      const float* nt_e  = nullt + le * D_;

      k_router<<<2048, 256, 0, stream>>>(xcur, rq_w, rk_w, sq, skv);
      k_coor<<<B_, 256, 0, stream>>>(sq,  scq,  NQ_);
      k_coor<<<B_, 256, 0, stream>>>(skv, sckv, NKV_);
      k_topk<<<B_, 512, 0, stream>>>(scq,  qi, qsc);
      k_topk<<<B_, 512, 0, stream>>>(sckv, ki, kvsc);
      k_gather_ln_bf<<<B_ * NQ_,  256, 0, stream>>>(xcur, qi, qg, qb, qnb);
      k_gather_ln_bf<<<B_ * NKV_, 256, 0, stream>>>(xcur, ki, cg, cb, ctxb);
      // q = qn @ wq : 4096x512x512 -> 4096 waves
      k_gemm_bf16<<<512, 256, 0, stream>>>(qnb, wqTe, qlin, B_ * NQ_, D_, D_);
      // kv = ctx @ wkv : 4096x512x1024 -> 8192 waves
      k_gemm_bf16<<<1024, 256, 0, stream>>>(ctxb, wkvTe, kvb, B_ * NKV_, D_, 2 * D_);
      k_qpost<<<EW_GRID_ROUT, EW_BLK, 0, stream>>>(qlin, qi, rot, qbf);
      k_kvpost<<<EW_GRID_ROUT, EW_BLK, 0, stream>>>(kvb, ki, kvsc, rot, kbbuf, vTbuf);
      // flash attention: 2048 waves
      k_attn<<<256, 256, 0, stream>>>(qbf, kbbuf, vTbuf, olinb);
      // o @ wo : 4096x512x512
      k_gemm_bf16<<<512, 256, 0, stream>>>(olinb, woTe, oproj, B_ * NQ_, D_, D_);
      k_scatter<<<EW_GRID_ROUT, EW_BLK, 0, stream>>>(oproj, qi, qsc, nt_e, acc);
    }

    k_combine<<<EW_GRID_BND, EW_BLK, 0, stream>>>(xcur, acc);
    // FFN
    k_ln_bf<<<B_ * N_, 256, 0, stream>>>(xcur, ffg + (size_t)l * D_, ffb + (size_t)l * D_, hlnb);
    k_gemm_bf16<<<2048, 256, 0, stream>>>(hlnb, ffw1T + (size_t)l * D_ * D_, h1, B_ * N_, D_, D_);
    k_bias_gelu_bf<<<EW_GRID_BND, EW_BLK, 0, stream>>>(h1, ffb1 + (size_t)l * D_, h1b);
    k_gemm_bf16<<<2048, 256, 0, stream>>>(h1b, ffw2T + (size_t)l * D_ * D_, h1, B_ * N_, D_, D_);
    k_bias_add_res<<<EW_GRID_BND, EW_BLK, 0, stream>>>(xcur, h1, ffb2 + (size_t)l * D_);
  }

  k_ln_f32<<<B_ * N_, 256, 0, stream>>>(xcur, outg, outb, (float*)d_out);
}
